// LibraKANMixer_56083682951294
// MI455X (gfx1250) — matrix operand order repaired and checked by reading, not compiled, not run on hardware
//
#include <hip/hip_runtime.h>
#include <math.h>

typedef __attribute__((ext_vector_type(16))) _Float16 v16h;
typedef __attribute__((ext_vector_type(8)))  _Float16 v8h;
typedef __attribute__((ext_vector_type(8)))  float    v8f;
typedef __attribute__((ext_vector_type(4)))  int      v4i;

// Problem constants (match reference)
constexpr int   B = 128, W = 1024, F = 2048;
constexpr float RHO = 0.3f, SPECTRAL_SCALE = 0.8f, ES_BETA = 6.0f, TAU = 0.001f;
constexpr float LN_EPS = 1e-5f, RMS_EPS = 1e-8f;

constexpr int NSPLIT = 4;   // F-dimension split for spectral partial sums
constexpr int BT     = 8;   // batch rows held in registers per spectral thread

// Workspace layout (in floats)
constexpr size_t XN_OFF    = 0;                                    // B*W f32
constexpr size_t THR_OFF   = XN_OFF + (size_t)B * W;               // F*W f32
constexpr size_t SPART_OFF = THR_OFF + (size_t)F * W;              // NSPLIT*B*W f32
constexpr size_t XNH_OFF   = SPART_OFF + (size_t)NSPLIT * B * W;   // 2*B*W halves
constexpr size_t WH_OFF    = XNH_OFF + (size_t)B * W;              // 2*W*W halves
constexpr size_t DENOM_OFF = WH_OFF + (size_t)W * W;               // 1 uint

// ---- gfx1250 async global->LDS path (guarded; falls back to plain copies) ----
#if defined(__has_builtin)
#if __has_builtin(__builtin_amdgcn_global_load_async_to_lds_b128)
#define HAVE_ASYNC_LDS 1
#endif
#endif

typedef __attribute__((address_space(1))) v4i gv4i;   // global int4
typedef __attribute__((address_space(3))) v4i lv4i;   // LDS int4

__device__ __forceinline__ void copy16_g2l(const _Float16* gsrc, _Float16* ldst) {
#if defined(HAVE_ASYNC_LDS)
    __builtin_amdgcn_global_load_async_to_lds_b128((gv4i*)gsrc, (lv4i*)ldst, 0, 0);
#else
    *(v8h*)ldst = *(const v8h*)gsrc;
#endif
}

__device__ __forceinline__ void wait_async_then_barrier() {
#if defined(HAVE_ASYNC_LDS)
#if __has_builtin(__builtin_amdgcn_s_wait_asynccnt)
    __builtin_amdgcn_s_wait_asynccnt(0);
#else
    asm volatile("s_wait_asynccnt 0x0" ::: "memory");
#endif
#endif
    __syncthreads();
}

__device__ __forceinline__ v16h cat8(v8h a, v8h b) {
    return __builtin_shufflevector(a, b, 0, 1, 2, 3, 4, 5, 6, 7,
                                         8, 9, 10, 11, 12, 13, 14, 15);
}

// ---------------------------------------------------------------------------
// 0) init: zero the max-abs accumulator
// ---------------------------------------------------------------------------
__global__ void init_kernel(float* ws) {
    if (threadIdx.x == 0) {
        ((unsigned int*)ws)[DENOM_OFF] = 0u;
    }
}

// ---------------------------------------------------------------------------
// 1) LayerNorm (one block per row) + fused f16 hi/lo split of x_n
// ---------------------------------------------------------------------------
__global__ void ln_kernel(const float* __restrict__ x,
                          const float* __restrict__ gamma,
                          const float* __restrict__ beta,
                          float* __restrict__ ws) {
    __shared__ float s_sum[256];
    __shared__ float s_sq[256];
    const int row = blockIdx.x;
    const int t   = threadIdx.x;
    const float* xr = x + (size_t)row * W;

    float sum = 0.f, sq = 0.f;
    #pragma unroll
    for (int j = 0; j < W / 256; ++j) {
        float v = xr[t + j * 256];
        sum += v;
        sq  += v * v;
    }
    s_sum[t] = sum; s_sq[t] = sq;
    __syncthreads();
    for (int off = 128; off > 0; off >>= 1) {
        if (t < off) { s_sum[t] += s_sum[t + off]; s_sq[t] += s_sq[t + off]; }
        __syncthreads();
    }
    const float mu  = s_sum[0] * (1.0f / W);
    const float var = s_sq[0] * (1.0f / W) - mu * mu;
    const float inv = rsqrtf(var + LN_EPS);

    float*     xn  = ws + XN_OFF + (size_t)row * W;
    _Float16*  xnh = (_Float16*)(ws + XNH_OFF) + (size_t)row * W;
    _Float16*  xnl = (_Float16*)(ws + XNH_OFF) + (size_t)B * W + (size_t)row * W;
    #pragma unroll
    for (int j = 0; j < W / 256; ++j) {
        const int c = t + j * 256;
        const float v = (xr[c] - mu) * inv * gamma[c] + beta[c];
        xn[c] = v;
        const _Float16 h = (_Float16)v;
        xnh[c] = h;
        xnl[c] = (_Float16)(v - (float)h);
    }
}

// ---------------------------------------------------------------------------
// 2) f16 hi/lo split of w_local (once; removes per-tile cvt work from GEMM)
// ---------------------------------------------------------------------------
__global__ void conv_w_kernel(const float* __restrict__ w_local,
                              float* __restrict__ ws) {
    const size_t i = (size_t)blockIdx.x * blockDim.x + threadIdx.x;
    if (i >= (size_t)W * W) return;
    _Float16* wh = (_Float16*)(ws + WH_OFF);
    _Float16* wl = wh + (size_t)W * W;
    const float v = w_local[i];
    const _Float16 h = (_Float16)v;
    wh[i] = h;
    wl[i] = (_Float16)(v - (float)h);
}

// ---------------------------------------------------------------------------
// 3) max|freq*scale| -> uint atomicMax on float bits (>=0, order-independent)
// ---------------------------------------------------------------------------
__global__ void maxabs_kernel(const float* __restrict__ freq, float* ws) {
    __shared__ float s_max[256];
    const size_t N = (size_t)F * W;
    float m = 0.f;
    for (size_t i = (size_t)blockIdx.x * blockDim.x + threadIdx.x; i < N;
         i += (size_t)gridDim.x * blockDim.x) {
        m = fmaxf(m, fabsf(freq[i] * SPECTRAL_SCALE));
    }
    s_max[threadIdx.x] = m;
    __syncthreads();
    for (int off = 128; off > 0; off >>= 1) {
        if (threadIdx.x < off)
            s_max[threadIdx.x] = fmaxf(s_max[threadIdx.x], s_max[threadIdx.x + off]);
        __syncthreads();
    }
    if (threadIdx.x == 0) {
        atomicMax((unsigned int*)ws + DENOM_OFF, __float_as_uint(s_max[0]));
    }
}

// ---------------------------------------------------------------------------
// 4) thresholded spectral weights: thr = sign(a*g)*relu(|a*g|-tau)*win
// ---------------------------------------------------------------------------
__global__ void thr_kernel(const float* __restrict__ freq,
                           const float* __restrict__ alpha,
                           const float* __restrict__ gate,
                           float* __restrict__ ws) {
    const size_t i = (size_t)blockIdx.x * blockDim.x + threadIdx.x;
    if (i >= (size_t)F * W) return;
    const float denom = __uint_as_float(((const unsigned int*)ws)[DENOM_OFF]);
    const float f = freq[i] * SPECTRAL_SCALE;
    float win;
    if (denom < 1e-8f) {
        win = 1.0f;
    } else {
        const float r = f / denom;
        win = __expf(-ES_BETA * r * r);
    }
    const float raw = alpha[i] * gate[i];
    const float mag = fmaxf(fabsf(raw) - TAU, 0.0f);
    const float sgn = (raw > 0.f) ? 1.f : ((raw < 0.f) ? -1.f : 0.f);
    ws[THR_OFF + i] = sgn * mag * win;
}

// ---------------------------------------------------------------------------
// 5) GEMM + bias + exact GELU, f16 WMMA with precomputed hi/lo operands.
//    Block = 8 waves, one 16-col slab (tn = blockIdx.x); wave handles the
//    16-row tile tm = waveId. B panel (16 x 1024 K, hi+lo halves, ~66 KB)
//    staged once per block in LDS via gfx1250 async global->LDS loads.
//    Inner loop: 4 global b128 (A) + 4 ds b128 (B) + 3 v_wmma per K=32.
// ---------------------------------------------------------------------------
constexpr int LPAD = 8;            // halves of row padding (bank stagger)
constexpr int LS   = W + LPAD;     // LDS row stride in halves

__global__ void gemm_gelu_kernel(const float* __restrict__ ws,
                                 const float* __restrict__ b_local,
                                 float* __restrict__ out) {
    __shared__ _Float16 lbh[16 * LS];
    __shared__ _Float16 lbl[16 * LS];

    const int t    = threadIdx.x;
    const int wave = t >> 5;            // 0..7  -> tm
    const int lane = t & 31;
    const int tn   = blockIdx.x;        // 0..63 -> 16-col slab
    const int m0   = wave * 16;
    const int n0   = tn * 16;
    const int mr   = lane & 15;         // A row / B col within tile
    const int hi   = lane >> 4;         // lane-half selector

    const _Float16* xnh = (const _Float16*)(ws + XNH_OFF);
    const _Float16* xnl = xnh + (size_t)B * W;
    const _Float16* wh  = (const _Float16*)(ws + WH_OFF);
    const _Float16* wl  = wh + (size_t)W * W;

    // ---- cooperative async fill of the B panel (rows n0..n0+15, all K) ----
    // 16 rows * 1024 halves = 2048 16-byte chunks per array; 8 per thread.
    #pragma unroll
    for (int i = 0; i < 8; ++i) {
        const int c  = t + 256 * i;
        const int r  = c >> 7;            // 0..15
        const int kc = (c & 127) * 8;     // halves within row
        copy16_g2l(wh + (size_t)(n0 + r) * W + kc, &lbh[r * LS + kc]);
        copy16_g2l(wl + (size_t)(n0 + r) * W + kc, &lbl[r * LS + kc]);
    }
    wait_async_then_barrier();

    v8f c = {};
    for (int k = 0; k < W; k += 32) {
        // A tile (16x32 f16): lane row m0+mr; halves e<8 -> K=k+hi*8+e,
        // e>=8 -> K=k+16+hi*8+(e-8): two contiguous 8-half runs.
        const _Float16* arh = xnh + (size_t)(m0 + mr) * W + k + hi * 8;
        const _Float16* arl = xnl + (size_t)(m0 + mr) * W + k + hi * 8;
        const v16h ahi = cat8(*(const v8h*)(arh), *(const v8h*)(arh + 16));
        const v16h alo = cat8(*(const v8h*)(arl), *(const v8h*)(arl + 16));

        // B tile (32x16 f16) from LDS: lane col n0+mr; same K order as A (two 8-half runs).
        const int boff = mr * LS + k + hi * 8;
        const v16h bhi = cat8(*(const v8h*)(lbh + boff), *(const v8h*)(lbh + boff + 16));
        const v16h blo = cat8(*(const v8h*)(lbl + boff), *(const v8h*)(lbl + boff + 16));

        // D += Ahi*Bhi + Ahi*Blo + Alo*Bhi  (drop lo*lo; ~1e-6 rel error)
        c = __builtin_amdgcn_wmma_f32_16x16x32_f16(false, ahi, false, bhi,
                                                   (short)0, c, false, false);
        c = __builtin_amdgcn_wmma_f32_16x16x32_f16(false, ahi, false, blo,
                                                   (short)0, c, false, false);
        c = __builtin_amdgcn_wmma_f32_16x16x32_f16(false, alo, false, bhi,
                                                   (short)0, c, false, false);
    }

    // Epilogue: C/D layout: VGPR i -> (M = m0 + i + hi*8, N = n0 + mr)
    const float bias = b_local[n0 + mr];
    #pragma unroll
    for (int i = 0; i < 8; ++i) {
        const int row = m0 + i + hi * 8;
        const float v = c[i] + bias;
        const float g = 0.5f * v * (1.0f + erff(v * 0.70710678118654752f));
        out[(size_t)row * W + (n0 + mr)] = g;
    }
}

// ---------------------------------------------------------------------------
// 6) Spectral partials: spart[z,b,w] = sum_{f in chunk z} sin(xn*f)*thr
//    BT=8 batch rows per thread; each (freq,thr) load amortized over 8 sins.
// ---------------------------------------------------------------------------
__global__ void spec_kernel(const float* __restrict__ freq,
                            float* __restrict__ ws) {
    const int w  = blockIdx.x * 256 + threadIdx.x;   // gridDim.x = W/256
    const int b0 = blockIdx.y * BT;                  // gridDim.y = B/BT
    const int z  = blockIdx.z;                       // gridDim.z = NSPLIT
    const int f0 = z * (F / NSPLIT);
    const int f1 = f0 + (F / NSPLIT);

    const float* xn  = ws + XN_OFF;
    const float* thr = ws + THR_OFF;

    float xv[BT];
    #pragma unroll
    for (int j = 0; j < BT; ++j) xv[j] = xn[(size_t)(b0 + j) * W + w];

    float acc[BT];
    #pragma unroll
    for (int j = 0; j < BT; ++j) acc[j] = 0.f;

    const float* fcol = freq + (size_t)f0 * W + w;
    const float* tcol = thr  + (size_t)f0 * W + w;
    for (int fi = f0; fi < f1; ++fi) {
        const float fv = (*fcol) * SPECTRAL_SCALE;
        const float tv = *tcol;
        __builtin_prefetch(fcol + 32 * W, 0, 0);     // global_prefetch_b8
        __builtin_prefetch(tcol + 32 * W, 0, 0);
        #pragma unroll
        for (int j = 0; j < BT; ++j)
            acc[j] += __sinf(xv[j] * fv) * tv;       // v_sin_f32 (TRANS)
        fcol += W;
        tcol += W;
    }

    float* spart = ws + SPART_OFF + (size_t)z * B * W;
    #pragma unroll
    for (int j = 0; j < BT; ++j)
        spart[(size_t)(b0 + j) * W + w] = acc[j];
}

// ---------------------------------------------------------------------------
// 7) Finalize: h_spec = sum_z spart; rms over W; out += RHO * h_spec / rms
// ---------------------------------------------------------------------------
__global__ void finalize_kernel(float* __restrict__ out,
                                const float* __restrict__ ws) {
    __shared__ float s_sq[256];
    const int b = blockIdx.x;
    const int t = threadIdx.x;
    const float* spart = ws + SPART_OFF;

    float hv[W / 256];
    float sq = 0.f;
    #pragma unroll
    for (int j = 0; j < W / 256; ++j) {
        const int w = t + j * 256;
        float s = 0.f;
        #pragma unroll
        for (int z = 0; z < NSPLIT; ++z)
            s += spart[((size_t)z * B + b) * W + w];
        hv[j] = s;
        sq += s * s;
    }
    s_sq[t] = sq;
    __syncthreads();
    for (int off = 128; off > 0; off >>= 1) {
        if (t < off) s_sq[t] += s_sq[t + off];
        __syncthreads();
    }
    const float inv_rms = rsqrtf(s_sq[0] * (1.0f / W) + RMS_EPS);

    #pragma unroll
    for (int j = 0; j < W / 256; ++j) {
        const int w = t + j * 256;
        out[(size_t)b * W + w] += RHO * hv[j] * inv_rms;
    }
}

// ---------------------------------------------------------------------------
extern "C" void kernel_launch(void* const* d_in, const int* in_sizes, int n_in,
                              void* d_out, int out_size, void* d_ws, size_t ws_size,
                              hipStream_t stream) {
    const float* x        = (const float*)d_in[0];
    const float* freq     = (const float*)d_in[1];
    const float* alpha    = (const float*)d_in[2];
    const float* gate     = (const float*)d_in[3];
    const float* ln_gamma = (const float*)d_in[4];
    const float* ln_beta  = (const float*)d_in[5];
    const float* w_local  = (const float*)d_in[6];
    const float* b_local  = (const float*)d_in[7];
    float*       out      = (float*)d_out;
    float*       ws       = (float*)d_ws;

    init_kernel<<<1, 32, 0, stream>>>(ws);
    ln_kernel<<<B, 256, 0, stream>>>(x, ln_gamma, ln_beta, ws);
    conv_w_kernel<<<(W * W) / 256, 256, 0, stream>>>(w_local, ws);
    maxabs_kernel<<<64, 256, 0, stream>>>(freq, ws);
    thr_kernel<<<(F * W) / 256, 256, 0, stream>>>(freq, alpha, gate, ws);
    gemm_gelu_kernel<<<64, 256, 0, stream>>>(ws, b_local, out);
    spec_kernel<<<dim3(W / 256, B / BT, NSPLIT), 256, 0, stream>>>(freq, ws);
    finalize_kernel<<<B, 256, 0, stream>>>(out, ws);
}
